// ProgramDecoder_35837207118391
// MI455X (gfx1250) — matrix operand order repaired and checked by reading, not compiled, not run on hardware
//
#include <hip/hip_runtime.h>
#include <hip/hip_bf16.h>
#include <cstdint>
#include <math.h>

typedef __bf16 bf16;
typedef __attribute__((ext_vector_type(16))) __bf16 v16bf;
typedef __attribute__((ext_vector_type(8)))  float  v8f;

#define B_   256
#define T_   64
#define H_   1024
#define E_   128
#define A_   128
#define O_   512
#define EPS_ 1e-5f
#define NSTEP 63

// ---------------------------------------------------------------------------
// Generic bf16 WMMA GEMM:  C(256 x N) = act( A(256 x K) @ W + bias )
// W supplied pre-transposed as Wt (N x K) bf16.
// Block: 128 threads = 4 waves. Block tile 32(M) x 256(N); wave tile 32x64
// (2 M-subtiles x 4 N-subtiles = 8 WMMAs per K-chunk; B fragments reused
// across both M-subtiles to halve weight traffic).
// No EXEC-divergence around the WMMAs: OOB waves are clamped to the last
// valid column tile (redundant compute) and only the store is predicated.
// ---------------------------------------------------------------------------
__global__ __launch_bounds__(128)
void gemm_bf16_wmma(const bf16* __restrict__ A, int lda,
                    const bf16* __restrict__ Wt,
                    const float* __restrict__ bias,
                    float* __restrict__ Cf, bf16* __restrict__ Cb,
                    int ldc, int N, int K, int do_tanh)
{
    // A tile (32 rows x 32 K) in contiguous k order: for row m the 32
    // K values are laid out as [k 0-31] in order so that
    // lane (l16, half) reads one contiguous, 32B-aligned 16-element run.
    __shared__ __align__(32) bf16 sA[32 * 32];

    const int tid  = threadIdx.x;
    const int wave = tid >> 5;
    const int lane = tid & 31;
    const int half = lane >> 4;      // 0: lanes 0-15, 1: lanes 16-31
    const int l16  = lane & 15;
    const int rowBase = blockIdx.y * 32;
    int colBase = blockIdx.x * 256 + wave * 64;
    const bool doStore = colBase < N;       // N is a multiple of 64
    if (colBase > N - 64) colBase = N - 64; // clamp -> uniform EXEC compute

    // B fragment base pointers (lane n holds column n; lanes 0-15 K 0-15,
    // lanes 16-31 K 16-31; contiguous in the transposed weights, 32B aligned)
    const bf16* bp0 = Wt + (size_t)(colBase +  0 + l16) * K + half * 16;
    const bf16* bp1 = Wt + (size_t)(colBase + 16 + l16) * K + half * 16;
    const bf16* bp2 = Wt + (size_t)(colBase + 32 + l16) * K + half * 16;
    const bf16* bp3 = Wt + (size_t)(colBase + 48 + l16) * K + half * 16;

    // Cooperative A-tile staging: 1024 bf16, 16 per thread at a stride of 8 (one 32B store),
    // stored in contiguous k order. c is a multiple of 8 -> single k-group.
    const int idx = tid * 8;
    const int r = idx >> 5, c = idx & 31;
    const int g = c >> 3;                       // 8-wide k-group: 0..3
    bf16* sdst = sA + r * 32 + c;
    const bf16* asrc = A + (size_t)(rowBase + r) * lda + c;

    // A fragment sources: one contiguous 32B run per lane per M-subtile
    const bf16* afrag0 = sA + l16 * 32 + half * 16;
    const bf16* afrag1 = sA + (16 + l16) * 32 + half * 16;

    v8f acc00 = {}, acc01 = {}, acc02 = {}, acc03 = {};
    v8f acc10 = {}, acc11 = {}, acc12 = {}, acc13 = {};

    for (int k0 = 0; k0 < K; k0 += 32) {
        *(v16bf*)sdst = *(const v16bf*)(asrc + k0);
        __syncthreads();

        v16bf a0 = *(const v16bf*)afrag0;
        v16bf a1 = *(const v16bf*)afrag1;
        v16bf bv0 = *(const v16bf*)(bp0 + k0);
        v16bf bv1 = *(const v16bf*)(bp1 + k0);
        v16bf bv2 = *(const v16bf*)(bp2 + k0);
        v16bf bv3 = *(const v16bf*)(bp3 + k0);
        __builtin_prefetch(bp0 + k0 + 32, 0, 3);   // WGP-scope, speculative
        __builtin_prefetch(bp2 + k0 + 32, 0, 3);

        acc00 = __builtin_amdgcn_wmma_f32_16x16x32_bf16(
                    false, a0, false, bv0, (short)0, acc00, false, false);
        acc10 = __builtin_amdgcn_wmma_f32_16x16x32_bf16(
                    false, a1, false, bv0, (short)0, acc10, false, false);
        acc01 = __builtin_amdgcn_wmma_f32_16x16x32_bf16(
                    false, a0, false, bv1, (short)0, acc01, false, false);
        acc11 = __builtin_amdgcn_wmma_f32_16x16x32_bf16(
                    false, a1, false, bv1, (short)0, acc11, false, false);
        acc02 = __builtin_amdgcn_wmma_f32_16x16x32_bf16(
                    false, a0, false, bv2, (short)0, acc02, false, false);
        acc12 = __builtin_amdgcn_wmma_f32_16x16x32_bf16(
                    false, a1, false, bv2, (short)0, acc12, false, false);
        acc03 = __builtin_amdgcn_wmma_f32_16x16x32_bf16(
                    false, a0, false, bv3, (short)0, acc03, false, false);
        acc13 = __builtin_amdgcn_wmma_f32_16x16x32_bf16(
                    false, a1, false, bv3, (short)0, acc13, false, false);

        __syncthreads();
    }

    if (!doStore) return;
    #pragma unroll
    for (int m = 0; m < 2; ++m) {
        #pragma unroll
        for (int s = 0; s < 4; ++s) {
            v8f acc;
            if (m == 0) acc = (s == 0) ? acc00 : (s == 1) ? acc01 : (s == 2) ? acc02 : acc03;
            else        acc = (s == 0) ? acc10 : (s == 1) ? acc11 : (s == 2) ? acc12 : acc13;
            int col = colBase + s * 16 + l16;
            float bv = bias ? bias[col] : 0.0f;
            #pragma unroll
            for (int q = 0; q < 8; ++q) {
                int row = rowBase + m * 16 + q + half * 8;  // vgpr q -> M=q / q+8
                float v = acc[q] + bv;
                if (do_tanh) v = tanhf(v);
                if (Cf) Cf[(size_t)row * ldc + col] = v;
                else    Cb[(size_t)row * ldc + col] = (bf16)v;
            }
        }
    }
}

// ---------------------------------------------------------------------------
// fp32 (K x N) -> bf16 transposed (N x K)
// ---------------------------------------------------------------------------
__global__ void conv_transpose(const float* __restrict__ W, bf16* __restrict__ Wt,
                               int K, int N)
{
    size_t idx = (size_t)blockIdx.x * 256 + threadIdx.x;
    if (idx >= (size_t)K * N) return;
    int n = (int)(idx / K);
    int k = (int)(idx - (size_t)n * K);
    Wt[idx] = (bf16)W[(size_t)k * N + n];
}

// input_emb -> bf16 copy; zero hx (fp32 + bf16)
__global__ void init_misc(const float* __restrict__ input_emb,
                          bf16* __restrict__ input_emb_bf,
                          float* __restrict__ hx, bf16* __restrict__ hxb)
{
    size_t idx = (size_t)blockIdx.x * 256 + threadIdx.x;
    if (idx >= (size_t)B_ * H_) return;
    input_emb_bf[idx] = (bf16)input_emb[idx];
    hx[idx] = 0.0f;
    hxb[idx] = (bf16)0.0f;
}

// Build atomic (B x 3E) bf16 from ground-truth tokens at step t
__global__ void gather_atomic(const int* __restrict__ ba, const int* __restrict__ bo1,
                              const int* __restrict__ bo2, int t,
                              const float* __restrict__ act_emb,
                              const float* __restrict__ obj_emb,
                              bf16* __restrict__ atomic)
{
    int b = blockIdx.x, tid = threadIdx.x;   // 128 threads
    int a  = ba [b * T_ + t];
    int o1 = bo1[b * T_ + t];
    int o2 = bo2[b * T_ + t];
    atomic[(size_t)b * 384 + tid]       = (bf16)act_emb[(size_t)a  * E_ + tid];
    atomic[(size_t)b * 384 + 128 + tid] = (bf16)obj_emb[(size_t)o1 * E_ + tid];
    atomic[(size_t)b * 384 + 256 + tid] = (bf16)obj_emb[(size_t)o2 * E_ + tid];
}

// ---------------------------------------------------------------------------
// Fused LayerNorm-GRU cell.  One block per batch row, 256 threads.
// LN over the full 3H = 3072 axis (matches reference _ln).
// ---------------------------------------------------------------------------
__device__ __forceinline__ float block_sum256(float v, float* sh)
{
    int tid = threadIdx.x;
    sh[tid] = v; __syncthreads();
    for (int s = 128; s > 0; s >>= 1) {
        if (tid < s) sh[tid] += sh[tid + s];
        __syncthreads();
    }
    float r = sh[0]; __syncthreads();
    return r;
}

__global__ __launch_bounds__(256)
void gru_cell(const float* __restrict__ gi, const float* __restrict__ gh,
              const float* __restrict__ g_ih, const float* __restrict__ be_ih,
              const float* __restrict__ g_hh, const float* __restrict__ be_hh,
              float* __restrict__ hx, bf16* __restrict__ hxb)
{
    __shared__ float sh[256];
    int b = blockIdx.x, tid = threadIdx.x;
    const float* gir = gi + (size_t)b * 3072;
    const float* ghr = gh + (size_t)b * 3072;
    float ci[12], ch[12];
    float si = 0, sqi = 0, s_h = 0, sqh = 0;
    #pragma unroll
    for (int i = 0; i < 12; ++i) {
        float a = gir[tid + i * 256]; ci[i] = a; si  += a; sqi += a * a;
        float c = ghr[tid + i * 256]; ch[i] = c; s_h += c; sqh += c * c;
    }
    float sum_i  = block_sum256(si,  sh);
    float sum_qi = block_sum256(sqi, sh);
    float sum_h  = block_sum256(s_h, sh);
    float sum_qh = block_sum256(sqh, sh);
    const float inv = 1.0f / 3072.0f;
    float mi = sum_i * inv,  vi = sum_qi * inv - mi * mi, rsi = rsqrtf(vi + EPS_);
    float mh = sum_h * inv,  vh = sum_qh * inv - mh * mh, rsh = rsqrtf(vh + EPS_);
    #pragma unroll
    for (int i = 0; i < 4; ++i) {
        int j = tid + i * 256;
        float ir  = (ci[i]     - mi) * rsi * g_ih[j]        + be_ih[j];
        float iz  = (ci[i + 4] - mi) * rsi * g_ih[j + 1024] + be_ih[j + 1024];
        float inn = (ci[i + 8] - mi) * rsi * g_ih[j + 2048] + be_ih[j + 2048];
        float hr  = (ch[i]     - mh) * rsh * g_hh[j]        + be_hh[j];
        float hz  = (ch[i + 4] - mh) * rsh * g_hh[j + 1024] + be_hh[j + 1024];
        float hn  = (ch[i + 8] - mh) * rsh * g_hh[j + 2048] + be_hh[j + 2048];
        float r = 1.0f / (1.0f + expf(-(ir + hr)));
        float z = 1.0f / (1.0f + expf(-(iz + hz)));
        float n = tanhf(inn + r * hn);
        float h0 = hx[(size_t)b * H_ + j];
        float hnew = (1.0f - z) * n + z * h0;
        hx [(size_t)b * H_ + j] = hnew;
        hxb[(size_t)b * H_ + j] = (bf16)hnew;
    }
}

// ---------------------------------------------------------------------------
// Action head: argmax + CE loss; builds bf16 concat inputs for object heads.
// Block = 128 threads (A = 128 logits).
// ---------------------------------------------------------------------------
__global__ __launch_bounds__(128)
void head_action(const float* __restrict__ logits, const int* __restrict__ ba,
                 int t, const float* __restrict__ act_emb,
                 const bf16* __restrict__ hidden,
                 bf16* __restrict__ o1_in, bf16* __restrict__ o2_in,
                 float* __restrict__ out)
{
    __shared__ float sv[128]; __shared__ int si[128];
    int b = blockIdx.x, tid = threadIdx.x;
    float v = logits[(size_t)b * A_ + tid];
    sv[tid] = v; si[tid] = tid; __syncthreads();
    for (int s = 64; s > 0; s >>= 1) {
        if (tid < s) {
            float ov = sv[tid + s]; int oi = si[tid + s];
            if (ov > sv[tid] || (ov == sv[tid] && oi < si[tid])) { sv[tid] = ov; si[tid] = oi; }
        }
        __syncthreads();
    }
    float mx = sv[0]; int amax = si[0]; __syncthreads();
    sv[tid] = expf(v - mx); __syncthreads();
    for (int s = 64; s > 0; s >>= 1) { if (tid < s) sv[tid] += sv[tid + s]; __syncthreads(); }
    if (tid == 0) {
        int tgt = ba[b * T_ + t + 1];
        out[((size_t)b * NSTEP + t) * 3 + 0] =
            logf(sv[0]) + mx - logits[(size_t)b * A_ + tgt];
        out[(size_t)B_ * NSTEP * 3 + (size_t)b * NSTEP + t] = (float)amax;
    }
    bf16 e = (bf16)act_emb[(size_t)amax * E_ + tid];
    o1_in[(size_t)b * 1152 + tid] = e;
    o2_in[(size_t)b * 1280 + tid] = e;
    #pragma unroll
    for (int i = 0; i < 8; ++i) {
        bf16 hv = hidden[(size_t)b * H_ + tid + i * 128];
        o1_in[(size_t)b * 1152 + 128 + tid + i * 128] = hv;
        o2_in[(size_t)b * 1280 + 256 + tid + i * 128] = hv;
    }
}

// ---------------------------------------------------------------------------
// Object head (O = 512 logits): argmax + CE; optional obj_emb gather into o2_in.
// Block = 256 threads, 2 logits per thread.
// ---------------------------------------------------------------------------
__global__ __launch_bounds__(256)
void head_obj(const float* __restrict__ logits, const int* __restrict__ bo,
              int t, const float* __restrict__ obj_emb,
              bf16* __restrict__ o2_in, float* __restrict__ out,
              int loss_k, size_t pred_base)
{
    __shared__ float sv[256]; __shared__ int si[256];
    int b = blockIdx.x, tid = threadIdx.x;
    float v0 = logits[(size_t)b * O_ + tid];
    float v1 = logits[(size_t)b * O_ + tid + 256];
    float mv; int mi_;
    if (v1 > v0) { mv = v1; mi_ = tid + 256; } else { mv = v0; mi_ = tid; }
    sv[tid] = mv; si[tid] = mi_; __syncthreads();
    for (int s = 128; s > 0; s >>= 1) {
        if (tid < s) {
            float ov = sv[tid + s]; int oi = si[tid + s];
            if (ov > sv[tid] || (ov == sv[tid] && oi < si[tid])) { sv[tid] = ov; si[tid] = oi; }
        }
        __syncthreads();
    }
    float mx = sv[0]; int amax = si[0]; __syncthreads();
    sv[tid] = expf(v0 - mx) + expf(v1 - mx); __syncthreads();
    for (int s = 128; s > 0; s >>= 1) { if (tid < s) sv[tid] += sv[tid + s]; __syncthreads(); }
    if (tid == 0) {
        int tgt = bo[b * T_ + t + 1];
        out[((size_t)b * NSTEP + t) * 3 + loss_k] =
            logf(sv[0]) + mx - logits[(size_t)b * O_ + tgt];
        out[pred_base + (size_t)b * NSTEP + t] = (float)amax;
    }
    if (o2_in && tid < 128)
        o2_in[(size_t)b * 1280 + 128 + tid] = (bf16)obj_emb[(size_t)amax * E_ + tid];
}

// ---------------------------------------------------------------------------
// Host side
// ---------------------------------------------------------------------------
extern "C" void kernel_launch(void* const* d_in, const int* in_sizes, int n_in,
                              void* d_out, int out_size, void* d_ws, size_t ws_size,
                              hipStream_t stream)
{
    const float* input_emb = (const float*)d_in[0];
    const int*   ba  = (const int*)d_in[1];
    const int*   bo1 = (const int*)d_in[2];
    const int*   bo2 = (const int*)d_in[3];
    const float* enc_W = (const float*)d_in[4];  const float* enc_b = (const float*)d_in[5];
    const float* act_emb = (const float*)d_in[6];const float* obj_emb = (const float*)d_in[7];
    const float* aa_W1 = (const float*)d_in[8];  const float* aa_b1 = (const float*)d_in[9];
    const float* aa_W2 = (const float*)d_in[10]; const float* aa_b2 = (const float*)d_in[11];
    const float* W_ih  = (const float*)d_in[12]; const float* b_ih  = (const float*)d_in[13];
    const float* W_hh  = (const float*)d_in[14]; const float* b_hh  = (const float*)d_in[15];
    const float* ln_ih_g = (const float*)d_in[16]; const float* ln_ih_b = (const float*)d_in[17];
    const float* ln_hh_g = (const float*)d_in[18]; const float* ln_hh_b = (const float*)d_in[19];
    const float* g2h_W1 = (const float*)d_in[20]; const float* g2h_b1 = (const float*)d_in[21];
    const float* g2h_W2 = (const float*)d_in[22]; const float* g2h_b2 = (const float*)d_in[23];
    const float* h2a_W1 = (const float*)d_in[24]; const float* h2a_b1 = (const float*)d_in[25];
    const float* h2a_W2 = (const float*)d_in[26]; const float* h2a_b2 = (const float*)d_in[27];
    const float* h2o1_W1 = (const float*)d_in[28]; const float* h2o1_b1 = (const float*)d_in[29];
    const float* h2o1_W2 = (const float*)d_in[30]; const float* h2o1_b2 = (const float*)d_in[31];
    const float* h2o2_W1 = (const float*)d_in[32]; const float* h2o2_b1 = (const float*)d_in[33];
    const float* h2o2_W2 = (const float*)d_in[34]; const float* h2o2_b2 = (const float*)d_in[35];

    float* out = (float*)d_out;

    // workspace bump allocator (256B aligned)
    char* p = (char*)d_ws;
    auto alloc = [&](size_t bytes) -> void* {
        void* r = (void*)p;
        p += (bytes + 255) & ~(size_t)255;
        return r;
    };

    // bf16 transposed weights (N x K)
    bf16* enc_Wt   = (bf16*)alloc((size_t)1024 * 1024 * 2);
    bf16* aa_W1t   = (bf16*)alloc((size_t)384  * 1024 * 2);
    bf16* aa_W2t   = (bf16*)alloc((size_t)1024 * 1024 * 2);
    bf16* W_iht    = (bf16*)alloc((size_t)2048 * 3072 * 2);
    bf16* W_hht    = (bf16*)alloc((size_t)1024 * 3072 * 2);
    bf16* g2h_W1t  = (bf16*)alloc((size_t)1024 * 1024 * 2);
    bf16* g2h_W2t  = (bf16*)alloc((size_t)1024 * 1024 * 2);
    bf16* h2a_W1t  = (bf16*)alloc((size_t)1024 * 1024 * 2);
    bf16* h2a_W2t  = (bf16*)alloc((size_t)1024 * 128  * 2);
    bf16* h2o1_W1t = (bf16*)alloc((size_t)1152 * 1024 * 2);
    bf16* h2o1_W2t = (bf16*)alloc((size_t)1024 * 512  * 2);
    bf16* h2o2_W1t = (bf16*)alloc((size_t)1280 * 1024 * 2);
    bf16* h2o2_W2t = (bf16*)alloc((size_t)1024 * 512  * 2);

    // activations
    bf16*  input_emb_bf = (bf16*)alloc((size_t)B_ * H_ * 2);
    bf16*  atomic_buf   = (bf16*)alloc((size_t)B_ * 384 * 2);
    bf16*  aa_h1        = (bf16*)alloc((size_t)B_ * H_ * 2);
    bf16*  xbuf         = (bf16*)alloc((size_t)B_ * 2048 * 2); // [aa | inp_enc]
    float* gi_raw       = (float*)alloc((size_t)B_ * 3072 * 4);
    float* gh_raw       = (float*)alloc((size_t)B_ * 3072 * 4);
    float* hx           = (float*)alloc((size_t)B_ * H_ * 4);
    bf16*  hxb          = (bf16*)alloc((size_t)B_ * H_ * 2);
    bf16*  g2h_t        = (bf16*)alloc((size_t)B_ * H_ * 2);
    bf16*  hidden       = (bf16*)alloc((size_t)B_ * H_ * 2);
    bf16*  t_a          = (bf16*)alloc((size_t)B_ * H_ * 2);
    float* a_logits     = (float*)alloc((size_t)B_ * A_ * 4);
    bf16*  o1_in        = (bf16*)alloc((size_t)B_ * 1152 * 2);
    bf16*  t_o1         = (bf16*)alloc((size_t)B_ * H_ * 2);
    float* o1_logits    = (float*)alloc((size_t)B_ * O_ * 4);
    bf16*  o2_in        = (bf16*)alloc((size_t)B_ * 1280 * 2);
    bf16*  t_o2         = (bf16*)alloc((size_t)B_ * H_ * 2);
    float* o2_logits    = (float*)alloc((size_t)B_ * O_ * 4);

    auto trans = [&](const float* W, bf16* Wt, int K, int N) {
        size_t total = (size_t)K * N;
        conv_transpose<<<dim3((unsigned)((total + 255) / 256)), 256, 0, stream>>>(W, Wt, K, N);
    };
    trans(enc_W,   enc_Wt,   1024, 1024);
    trans(aa_W1,   aa_W1t,   384,  1024);
    trans(aa_W2,   aa_W2t,   1024, 1024);
    trans(W_ih,    W_iht,    2048, 3072);
    trans(W_hh,    W_hht,    1024, 3072);
    trans(g2h_W1,  g2h_W1t,  1024, 1024);
    trans(g2h_W2,  g2h_W2t,  1024, 1024);
    trans(h2a_W1,  h2a_W1t,  1024, 1024);
    trans(h2a_W2,  h2a_W2t,  1024, 128);
    trans(h2o1_W1, h2o1_W1t, 1152, 1024);
    trans(h2o1_W2, h2o1_W2t, 1024, 512);
    trans(h2o2_W1, h2o2_W1t, 1280, 1024);
    trans(h2o2_W2, h2o2_W2t, 1024, 512);

    init_misc<<<dim3((B_ * H_ + 255) / 256), 256, 0, stream>>>(
        input_emb, input_emb_bf, hx, hxb);

    auto gemm = [&](const bf16* Ap, int lda, const bf16* Wt, const float* bias,
                    float* Cf, bf16* Cb, int ldc, int N, int K, int do_tanh) {
        dim3 grid((N + 255) / 256, B_ / 32);
        gemm_bf16_wmma<<<grid, 128, 0, stream>>>(Ap, lda, Wt, bias, Cf, Cb,
                                                 ldc, N, K, do_tanh);
    };

    // inp_enc = tanh(input_emb @ enc_W + enc_b) -> x[:, 1024:2048] (bf16)
    gemm(input_emb_bf, 1024, enc_Wt, enc_b, nullptr, xbuf + 1024, 2048, 1024, 1024, 1);

    const size_t pred_a  = (size_t)B_ * NSTEP * 3;
    const size_t pred_o1 = pred_a + (size_t)B_ * NSTEP;
    const size_t pred_o2 = pred_o1 + (size_t)B_ * NSTEP;

    for (int t = 0; t < NSTEP; ++t) {
        // atomic action embedding (teacher forced tokens at step t)
        gather_atomic<<<B_, 128, 0, stream>>>(ba, bo1, bo2, t, act_emb, obj_emb,
                                              atomic_buf);
        // aa = tanh(tanh(atomic @ aa_W1 + b1) @ aa_W2 + b2) -> x[:, 0:1024]
        gemm(atomic_buf, 384, aa_W1t, aa_b1, nullptr, aa_h1, 1024, 1024, 384, 1);
        gemm(aa_h1, 1024, aa_W2t, aa_b2, nullptr, xbuf, 2048, 1024, 1024, 1);
        // GRU gate pre-activations
        gemm(xbuf, 2048, W_iht, b_ih, gi_raw, nullptr, 3072, 3072, 2048, 0);
        gemm(hxb, 1024, W_hht, b_hh, gh_raw, nullptr, 3072, 3072, 1024, 0);
        // fused LayerNorm + GRU update
        gru_cell<<<B_, 256, 0, stream>>>(gi_raw, gh_raw, ln_ih_g, ln_ih_b,
                                         ln_hh_g, ln_hh_b, hx, hxb);
        // hidden = tanh(tanh(hx @ g2h_W1 + b1) @ g2h_W2 + b2)
        gemm(hxb, 1024, g2h_W1t, g2h_b1, nullptr, g2h_t, 1024, 1024, 1024, 1);
        gemm(g2h_t, 1024, g2h_W2t, g2h_b2, nullptr, hidden, 1024, 1024, 1024, 1);
        // action head
        gemm(hidden, 1024, h2a_W1t, h2a_b1, nullptr, t_a, 1024, 1024, 1024, 1);
        gemm(t_a, 1024, h2a_W2t, h2a_b2, a_logits, nullptr, 128, 128, 1024, 0);
        head_action<<<B_, 128, 0, stream>>>(a_logits, ba, t, act_emb, hidden,
                                            o1_in, o2_in, out);
        // object1 head
        gemm(o1_in, 1152, h2o1_W1t, h2o1_b1, nullptr, t_o1, 1024, 1024, 1152, 1);
        gemm(t_o1, 1024, h2o1_W2t, h2o1_b2, o1_logits, nullptr, 512, 512, 1024, 0);
        head_obj<<<B_, 256, 0, stream>>>(o1_logits, bo1, t, obj_emb, o2_in, out,
                                         1, pred_o1);
        // object2 head
        gemm(o2_in, 1280, h2o2_W1t, h2o2_b1, nullptr, t_o2, 1024, 1024, 1280, 1);
        gemm(t_o2, 1024, h2o2_W2t, h2o2_b2, o2_logits, nullptr, 512, 512, 1024, 0);
        head_obj<<<B_, 256, 0, stream>>>(o2_logits, bo2, t, obj_emb, nullptr, out,
                                         2, pred_o2);
    }
    (void)in_sizes; (void)n_in; (void)out_size; (void)ws_size;
}
